// SlidingWindowAttention_9277129359803
// MI455X (gfx1250) — matrix operand order repaired and checked by reading, not compiled, not run on hardware
//
#include <hip/hip_runtime.h>

typedef __bf16 bf16_t;
typedef bf16_t v16bf __attribute__((ext_vector_type(16)));
typedef bf16_t v8bf  __attribute__((ext_vector_type(8)));
typedef float  v8f   __attribute__((ext_vector_type(8)));

#define BB   2
#define LL   2048
#define DD   1024
#define HH   16
#define EE   64
#define WIN  256

// ---------------- fragment helpers (CDNA5 WMMA bf16 layouts, wave32) -------
// A 16x32 bf16: lane holds row lane%16; K = {hi*8+0..7, hi*8+16..23}
static __device__ __forceinline__ v16bf ld_afrag(const bf16_t* rowbase, int k0, int hi) {
  v8bf a = *(const v8bf*)(rowbase + k0 + hi * 8);
  v8bf b = *(const v8bf*)(rowbase + k0 + hi * 8 + 16);
  return __builtin_shufflevector(a, b, 0,1,2,3,4,5,6,7,8,9,10,11,12,13,14,15);
}
// B 32x16 bf16: lane holds col lane%16; K = {hi*8+0..7, hi*8+16..23}
static __device__ __forceinline__ v16bf ld_bfrag(const bf16_t* colbase, int k0, int hi) {
  v8bf a = *(const v8bf*)(colbase + k0 + hi * 8);
  v8bf b = *(const v8bf*)(colbase + k0 + hi * 8 + 16);
  return __builtin_shufflevector(a, b, 0,1,2,3,4,5,6,7,8,9,10,11,12,13,14,15);
}
static __device__ __forceinline__ v8f wmma_bf16(v16bf a, v16bf b, v8f c) {
  return __builtin_amdgcn_wmma_f32_16x16x32_bf16(false, a, false, b, (short)0, c,
                                                 false, false);
}

// ---------------- fp32 -> bf16 convert -------------------------------------
__global__ __launch_bounds__(256) void swa_cvt_bf16(const float* __restrict__ in,
                                                    bf16_t* __restrict__ out, int n) {
  int i = blockIdx.x * blockDim.x + threadIdx.x;
  int stride = gridDim.x * blockDim.x;
  for (; i < n; i += stride) out[i] = (bf16_t)in[i];
}

// ---------------- QKV GEMM: [4096,1024] x [3072,1024]^T + bias -------------
// one wave computes a 32x64 tile (2 M-blocks x 4 N-blocks, 8 WMMA/K-step);
// scatters to Q[B,H,L,E], K[B,H,L,E], V^T[B,H,E,L]
__global__ __launch_bounds__(256) void swa_qkv_gemm(const bf16_t* __restrict__ xb,
                                                    const bf16_t* __restrict__ wb,
                                                    const float* __restrict__ bias,
                                                    bf16_t* __restrict__ q,
                                                    bf16_t* __restrict__ k,
                                                    bf16_t* __restrict__ vt) {
  const int lane = threadIdx.x & 31;
  const int hi = lane >> 4, lm = lane & 15;
  const int wave = blockIdx.x * (blockDim.x >> 5) + (threadIdx.x >> 5);
  const int NT4 = (3 * DD) / 64;                       // 48
  const int mt = wave / NT4, nt = wave % NT4;
  const int m0 = mt * 32, n0 = nt * 64;

  const bf16_t* arow0 = xb + (size_t)(m0 + lm) * DD;
  const bf16_t* arow1 = xb + (size_t)(m0 + 16 + lm) * DD;
  const bf16_t* brow0 = wb + (size_t)(n0 + 0 * 16 + lm) * DD;
  const bf16_t* brow1 = wb + (size_t)(n0 + 1 * 16 + lm) * DD;
  const bf16_t* brow2 = wb + (size_t)(n0 + 2 * 16 + lm) * DD;
  const bf16_t* brow3 = wb + (size_t)(n0 + 3 * 16 + lm) * DD;

  v8f acc[2][4] = {};
  for (int k0 = 0; k0 < DD; k0 += 32) {
    v16bf af0 = ld_afrag(arow0, k0, hi);
    v16bf af1 = ld_afrag(arow1, k0, hi);
    v16bf bf0 = ld_bfrag(brow0, k0, hi);
    acc[0][0] = wmma_bf16(af0, bf0, acc[0][0]);
    acc[1][0] = wmma_bf16(af1, bf0, acc[1][0]);
    v16bf bf1 = ld_bfrag(brow1, k0, hi);
    acc[0][1] = wmma_bf16(af0, bf1, acc[0][1]);
    acc[1][1] = wmma_bf16(af1, bf1, acc[1][1]);
    v16bf bf2 = ld_bfrag(brow2, k0, hi);
    acc[0][2] = wmma_bf16(af0, bf2, acc[0][2]);
    acc[1][2] = wmma_bf16(af1, bf2, acc[1][2]);
    v16bf bf3 = ld_bfrag(brow3, k0, hi);
    acc[0][3] = wmma_bf16(af0, bf3, acc[0][3]);
    acc[1][3] = wmma_bf16(af1, bf3, acc[1][3]);
  }

#pragma unroll
  for (int j = 0; j < 4; ++j) {
    const int n = n0 + j * 16 + lm;
    const float bv = bias[n];
    const int which = n >> 10;            // 0=q 1=k 2=v
    const int d = n & (DD - 1);
    const int h = d >> 6, e = d & 63;
#pragma unroll
    for (int mb = 0; mb < 2; ++mb) {
#pragma unroll
      for (int r = 0; r < 8; ++r) {
        const int m = m0 + mb * 16 + r + hi * 8;
        const int bi = m >> 11;           // / LL
        const int l = m & (LL - 1);
        const bf16_t o = (bf16_t)(acc[mb][j][r] + bv);
        if (which == 0)
          q[((((size_t)bi * HH + h) * LL + l) << 6) + e] = o;
        else if (which == 1)
          k[((((size_t)bi * HH + h) * LL + l) << 6) + e] = o;
        else
          vt[(((size_t)bi * HH + h) * EE + e) * LL + l] = o;
      }
    }
  }
}

// ---------------- sliding-window flash attention ---------------------------
// one wave per (b,h, 16-query tile); flash online softmax; P transposed via LDS
__global__ __launch_bounds__(256) void swa_attn(const bf16_t* __restrict__ q,
                                                const bf16_t* __restrict__ k,
                                                const bf16_t* __restrict__ vt,
                                                bf16_t* __restrict__ ao) {
  __shared__ bf16_t pbuf[8][16 * 40];     // per-wave 16x32 P tile, padded rows
  const int lane = threadIdx.x & 31;
  const int hi = lane >> 4, lm = lane & 15;
  const int wv = threadIdx.x >> 5;
  const int gw = blockIdx.x * 8 + wv;     // 4096 waves total
  const int it = gw & 127;                // L/16
  const int bh = gw >> 7;                 // b*16 + h
  const int b = bh >> 4, h = bh & 15;
  const int i0 = it * 16;

  const bf16_t* qbase = q + (size_t)bh * LL * EE;
  const bf16_t* kbase = k + (size_t)bh * LL * EE;
  const bf16_t* vbase = vt + (size_t)bh * EE * LL;

  const bf16_t* qrow = qbase + (size_t)(i0 + lm) * EE;
  const v16bf qa0 = ld_afrag(qrow, 0, hi);
  const v16bf qa1 = ld_afrag(qrow, 32, hi);

  v8f acc[4] = {};
  float mrun[8], lrun[8], scr[8];
#pragma unroll
  for (int r = 0; r < 8; ++r) { mrun[r] = -1e30f; lrun[r] = 0.0f; }

  int js = i0 - (WIN - 1);
  if (js < 0) js = 0;
  js &= ~31;

  for (int j0 = js; j0 <= i0 + 15; j0 += 32) {
    // ---- scores S = (Q K^T) * 1/sqrt(E), two 16-key column blocks ----
    v8f s0, s1;
    {
      int j = j0 + lm; int jr = (j < LL - 1) ? j : (LL - 1);
      const bf16_t* krow = kbase + (size_t)jr * EE;
      v8f t = {};
      t = wmma_bf16(qa0, ld_bfrag(krow, 0, hi), t);
      t = wmma_bf16(qa1, ld_bfrag(krow, 32, hi), t);
      s0 = t * 0.125f;
    }
    {
      int j = j0 + 16 + lm; int jr = (j < LL - 1) ? j : (LL - 1);
      const bf16_t* krow = kbase + (size_t)jr * EE;
      v8f t = {};
      t = wmma_bf16(qa0, ld_bfrag(krow, 0, hi), t);
      t = wmma_bf16(qa1, ld_bfrag(krow, 32, hi), t);
      s1 = t * 0.125f;
    }

    // ---- online softmax stats per row (rows live across 16-lane halves) ----
#pragma unroll
    for (int r = 0; r < 8; ++r) {
      const int m = i0 + r + hi * 8;
      const int jc0 = j0 + lm;
      const int jc1 = j0 + 16 + lm;
      const bool mk0 = (jc0 > m) || ((m - jc0) >= WIN);
      const bool mk1 = (jc1 > m) || ((m - jc1) >= WIN);
      float a0 = mk0 ? -1e30f : s0[r];
      float a1 = mk1 ? -1e30f : s1[r];
      float rmax = fmaxf(a0, a1);
      rmax = fmaxf(rmax, __shfl_xor(rmax, 1));
      rmax = fmaxf(rmax, __shfl_xor(rmax, 2));
      rmax = fmaxf(rmax, __shfl_xor(rmax, 4));
      rmax = fmaxf(rmax, __shfl_xor(rmax, 8));
      const float mnew = fmaxf(mrun[r], rmax);
      const float sc = __expf(mrun[r] - mnew);
      scr[r] = sc;
      mrun[r] = mnew;
      const float p0 = mk0 ? 0.0f : __expf(s0[r] - mnew);
      const float p1 = mk1 ? 0.0f : __expf(s1[r] - mnew);
      float ps = p0 + p1;
      ps += __shfl_xor(ps, 1);
      ps += __shfl_xor(ps, 2);
      ps += __shfl_xor(ps, 4);
      ps += __shfl_xor(ps, 8);
      lrun[r] = lrun[r] * sc + ps;
      const int row = r + hi * 8;
      pbuf[wv][row * 40 + lm] = (bf16_t)p0;
      pbuf[wv][row * 40 + 16 + lm] = (bf16_t)p1;
    }
    asm volatile("s_wait_dscnt 0x0" ::: "memory");

    // ---- rescale running accumulator ----
#pragma unroll
    for (int eb = 0; eb < 4; ++eb)
#pragma unroll
      for (int r = 0; r < 8; ++r) acc[eb][r] *= scr[r];

    // ---- O += P @ V  (A-frag of P from LDS, B-frag of V^T contiguous) ----
    const v16bf pf = ld_afrag(&pbuf[wv][lm * 40], 0, hi);
#pragma unroll
    for (int eb = 0; eb < 4; ++eb) {
      const bf16_t* vcol = vbase + (size_t)(eb * 16 + lm) * LL;
      acc[eb] = wmma_bf16(pf, ld_bfrag(vcol, j0, hi), acc[eb]);
    }
  }

  // ---- epilogue: normalize, store [B, L, H, E] as bf16 ----
#pragma unroll
  for (int r = 0; r < 8; ++r) {
    const int m = i0 + r + hi * 8;
    const float inv = 1.0f / lrun[r];
#pragma unroll
    for (int eb = 0; eb < 4; ++eb) {
      const int e = eb * 16 + lm;
      ao[(((size_t)b * LL + m) * HH + h) * EE + e] = (bf16_t)(acc[eb][r] * inv);
    }
  }
}

// ---------------- output projection: [4096,1024] x [1024,1024]^T + bias ----
// one wave computes a 32x64 tile (8 WMMA per K-step)
__global__ __launch_bounds__(256) void swa_proj_gemm(const bf16_t* __restrict__ ab,
                                                     const bf16_t* __restrict__ wb,
                                                     const float* __restrict__ bias,
                                                     float* __restrict__ out) {
  const int lane = threadIdx.x & 31;
  const int hi = lane >> 4, lm = lane & 15;
  const int wave = blockIdx.x * (blockDim.x >> 5) + (threadIdx.x >> 5);
  const int NT4 = DD / 64;                             // 16
  const int mt = wave / NT4, nt = wave % NT4;
  const int m0 = mt * 32, n0 = nt * 64;

  const bf16_t* arow0 = ab + (size_t)(m0 + lm) * DD;
  const bf16_t* arow1 = ab + (size_t)(m0 + 16 + lm) * DD;
  const bf16_t* brow0 = wb + (size_t)(n0 + 0 * 16 + lm) * DD;
  const bf16_t* brow1 = wb + (size_t)(n0 + 1 * 16 + lm) * DD;
  const bf16_t* brow2 = wb + (size_t)(n0 + 2 * 16 + lm) * DD;
  const bf16_t* brow3 = wb + (size_t)(n0 + 3 * 16 + lm) * DD;

  v8f acc[2][4] = {};
  for (int k0 = 0; k0 < DD; k0 += 32) {
    v16bf af0 = ld_afrag(arow0, k0, hi);
    v16bf af1 = ld_afrag(arow1, k0, hi);
    v16bf bf0 = ld_bfrag(brow0, k0, hi);
    acc[0][0] = wmma_bf16(af0, bf0, acc[0][0]);
    acc[1][0] = wmma_bf16(af1, bf0, acc[1][0]);
    v16bf bf1 = ld_bfrag(brow1, k0, hi);
    acc[0][1] = wmma_bf16(af0, bf1, acc[0][1]);
    acc[1][1] = wmma_bf16(af1, bf1, acc[1][1]);
    v16bf bf2 = ld_bfrag(brow2, k0, hi);
    acc[0][2] = wmma_bf16(af0, bf2, acc[0][2]);
    acc[1][2] = wmma_bf16(af1, bf2, acc[1][2]);
    v16bf bf3 = ld_bfrag(brow3, k0, hi);
    acc[0][3] = wmma_bf16(af0, bf3, acc[0][3]);
    acc[1][3] = wmma_bf16(af1, bf3, acc[1][3]);
  }

#pragma unroll
  for (int j = 0; j < 4; ++j) {
    const int n = n0 + j * 16 + lm;
    const float bv = bias[n];
#pragma unroll
    for (int mb = 0; mb < 2; ++mb) {
#pragma unroll
      for (int r = 0; r < 8; ++r) {
        const int m = m0 + mb * 16 + r + hi * 8;
        out[(size_t)m * DD + n] = acc[mb][j][r] + bv;
      }
    }
  }
}

// ---------------- host side -------------------------------------------------
extern "C" void kernel_launch(void* const* d_in, const int* in_sizes, int n_in,
                              void* d_out, int out_size, void* d_ws, size_t ws_size,
                              hipStream_t stream) {
  const float* x      = (const float*)d_in[0];   // [B,L,D]
  const float* w_qkv  = (const float*)d_in[1];   // [3D,D]
  const float* b_qkv  = (const float*)d_in[2];   // [3D]
  const float* w_proj = (const float*)d_in[3];   // [D,D]
  const float* b_proj = (const float*)d_in[4];   // [D]
  float* out = (float*)d_out;

  char* ws = (char*)d_ws;
  constexpr size_t N_X   = (size_t)BB * LL * DD;        // 4M
  constexpr size_t N_WQ  = (size_t)3 * DD * DD;         // 3M
  constexpr size_t N_WP  = (size_t)DD * DD;             // 1M
  constexpr size_t N_QKV = (size_t)BB * HH * LL * EE;   // 4M per tensor

  constexpr size_t OFF_XB   = 0;
  constexpr size_t OFF_WQKV = OFF_XB + N_X * 2;
  constexpr size_t OFF_WPRJ = OFF_WQKV + N_WQ * 2;
  constexpr size_t OFF_Q    = OFF_WPRJ + N_WP * 2;
  constexpr size_t OFF_K    = OFF_Q + N_QKV * 2;
  constexpr size_t OFF_VT   = OFF_K + N_QKV * 2;
  constexpr size_t OFF_ATTN = OFF_VT + N_QKV * 2 + 128; // pad for OOB tail reads

  bf16_t* xb   = (bf16_t*)(ws + OFF_XB);
  bf16_t* wqb  = (bf16_t*)(ws + OFF_WQKV);
  bf16_t* wpb  = (bf16_t*)(ws + OFF_WPRJ);
  bf16_t* qb   = (bf16_t*)(ws + OFF_Q);
  bf16_t* kb   = (bf16_t*)(ws + OFF_K);
  bf16_t* vtb  = (bf16_t*)(ws + OFF_VT);
  bf16_t* atb  = (bf16_t*)(ws + OFF_ATTN);

  swa_cvt_bf16<<<2048, 256, 0, stream>>>(x, xb, (int)N_X);
  swa_cvt_bf16<<<2048, 256, 0, stream>>>(w_qkv, wqb, (int)N_WQ);
  swa_cvt_bf16<<<1024, 256, 0, stream>>>(w_proj, wpb, (int)N_WP);

  // (4096/32) M-tiles * (3072/64) N-groups = 6144 waves -> 768 blocks of 8 waves
  swa_qkv_gemm<<<768, 256, 0, stream>>>(xb, wqb, b_qkv, qb, kb, vtb);

  // B*H*(L/16) = 4096 waves -> 512 blocks
  swa_attn<<<512, 256, 0, stream>>>(qb, kb, vtb, atb);

  // (4096/32) M-tiles * (1024/64) N-groups = 2048 waves -> 256 blocks
  swa_proj_gemm<<<256, 256, 0, stream>>>(atb, wpb, b_proj, out);
}
